// Attention_50122268344415
// MI455X (gfx1250) — matrix operand order repaired and checked by reading, not compiled, not run on hardware
//
#include <hip/hip_runtime.h>

// ---------------------------------------------------------------------------
// Fused attention for MI455X (gfx1250, wave32, WMMA).
//   q = x·Wq + bq ; k = x·Wk + bk ; v = x·Wv + bv
//   out = softmax(q·kᵀ / sqrt(dk)) · v
// bf16 WMMA (v_wmma_f32_16x16x32_bf16) for all three GEMM stages, fp32
// accumulate. Fragments are loaded in the exact ISA lane layout (two b128
// loads per fragment). Inner k-loops are manually ping-ponged (unroll-by-2
// with two named buffer sets) so loads land directly in the registers the
// next WMMA group consumes — no rotation moves — and each wave overlaps its
// own loads with WMMA execution. Scores for a 16-query tile live entirely in
// LDS (128 KB of the WGP's 320 KB): softmax is a simple wave32 reduction.
// ---------------------------------------------------------------------------

#define D_MODEL 1024
#define DKDIM   1024
#define DVDIM   1024
#define BATCH   8
#define SEQ     2048
#define MTOT    (BATCH * SEQ)   // 16384 rows total

typedef __attribute__((ext_vector_type(16))) __bf16 v16bf;
typedef __attribute__((ext_vector_type(8)))  __bf16 v8bf;
typedef __attribute__((ext_vector_type(8)))  float  v8f;

// Build a 16-element bf16 fragment from two contiguous 16-byte chunks.
static __device__ __forceinline__ v16bf load_frag(const __bf16* p0, const __bf16* p1) {
  v8bf lo = *(const v8bf*)p0;
  v8bf hi = *(const v8bf*)p1;
  return __builtin_shufflevector(lo, hi, 0,1,2,3,4,5,6,7,8,9,10,11,12,13,14,15);
}

// Build a bf16 fragment from two runs of 8 fp32 values (LDS probs -> A frag).
static __device__ __forceinline__ v16bf frag_from_f32(const float* p0, const float* p1) {
  v16bf r;
#pragma unroll
  for (int i = 0; i < 8; ++i) { r[i] = (__bf16)p0[i]; r[i + 8] = (__bf16)p1[i]; }
  return r;
}

static __device__ __forceinline__ v8f wmma_bf16(v16bf a, v16bf b, v8f c) {
  return __builtin_amdgcn_wmma_f32_16x16x32_bf16(false, a, false, b, (short)0, c,
                                                 false, false);
}

static __device__ __forceinline__ void zero8(v8f& v) {
#pragma unroll
  for (int j = 0; j < 8; ++j) v[j] = 0.0f;
}

// ---- fragment loaders ------------------------------------------------------
// A (16x32 bf16, row-major): lane row r applied by caller; elems 0..7 =
// K[k+hi*8 ..], elems 8..15 = K[k+16+hi*8 ..].
template <int MT>
static __device__ __forceinline__ void loadA(v16bf a[MT], const __bf16* arow,
                                             size_t astride, int k, int hi) {
#pragma unroll
  for (int i = 0; i < MT; ++i) {
    const __bf16* p = arow + (size_t)i * astride + k + hi * 8;
    a[i] = load_frag(p, p + 16 - hi * 8 + hi * 8);   // p and p+16 same lane row
  }
}
// B (32x16 bf16, K-contiguous per lane column): elems 0..7 = K[k+hi*8 ..],
// elems 8..15 = K[k+16+hi*8 ..].
static __device__ __forceinline__ void loadB(v16bf b[4], const __bf16* bbase,
                                             size_t btile, int k, int hi) {
#pragma unroll
  for (int t = 0; t < 4; ++t) {
    const __bf16* p = bbase + (size_t)t * btile + k + hi * 8;
    b[t] = load_frag(p, p + 16);
  }
}

template <int MT>
static __device__ __forceinline__ void mma_group(v8f acc[MT][4], const v16bf a[MT],
                                                 const v16bf b[4]) {
#pragma unroll
  for (int i = 0; i < MT; ++i)
#pragma unroll
    for (int t = 0; t < 4; ++t) acc[i][t] = wmma_bf16(a[i], b[t], acc[i][t]);
}

// ---------------------------------------------------------------------------
// Ping-pong pipelined (MT*16)xK x Kx64 accumulate, all operands from global
// bf16. kdim must be a multiple of 64. Buffer set A is consumed while set B
// loads, and vice versa — loads write directly into the consuming registers.
// ---------------------------------------------------------------------------
template <int MT>
static __device__ __forceinline__ void gemm_pingpong(
    const __bf16* __restrict__ arow, size_t astride,
    const __bf16* __restrict__ bbase, size_t btile,
    int hi, int kdim, v8f acc[MT][4]) {
  v16bf aA[MT], bA[4], aB[MT], bB[4];
  loadA<MT>(aA, arow, astride, 0, hi);
  loadB(bA, bbase, btile, 0, hi);
  for (int k0 = 0; k0 < kdim; k0 += 64) {
    loadA<MT>(aB, arow, astride, k0 + 32, hi);
    loadB(bB, bbase, btile, k0 + 32, hi);
    mma_group<MT>(acc, aA, bA);
    const int k2 = (k0 + 64 < kdim) ? k0 + 64 : 0;   // wrap keeps loads defined
    loadA<MT>(aA, arow, astride, k2, hi);
    loadB(bA, bbase, btile, k2, hi);
    mma_group<MT>(acc, aB, bB);
  }
}

// Same, MT=1, but A fragments come from an LDS fp32 row (probs) with cvt.
static __device__ __forceinline__ void gemm_pv_pingpong(
    const float* __restrict__ prow,
    const __bf16* __restrict__ bbase, size_t btile,
    int hi, int kdim, v8f acc[4]) {
  v16bf aA, bA[4], aB, bB[4];
  aA = frag_from_f32(prow + hi * 8, prow + 16 + hi * 8);
  loadB(bA, bbase, btile, 0, hi);
  for (int k0 = 0; k0 < kdim; k0 += 64) {
    aB = frag_from_f32(prow + k0 + 32 + hi * 8, prow + k0 + 48 + hi * 8);
    loadB(bB, bbase, btile, k0 + 32, hi);
#pragma unroll
    for (int t = 0; t < 4; ++t) acc[t] = wmma_bf16(aA, bA[t], acc[t]);
    const int k2 = (k0 + 64 < kdim) ? k0 + 64 : 0;
    aA = frag_from_f32(prow + k2 + hi * 8, prow + k2 + 16 + hi * 8);
    loadB(bA, bbase, btile, k2, hi);
#pragma unroll
    for (int t = 0; t < 4; ++t) acc[t] = wmma_bf16(aB, bB[t], acc[t]);
  }
}

// ---------------------------------------------------------------------------
// Kernel 1a: fp32 -> bf16 convert (x -> xh)
// ---------------------------------------------------------------------------
__global__ void k_cvt_x(const float* __restrict__ x, __bf16* __restrict__ xh, int n) {
  int i = blockIdx.x * blockDim.x + threadIdx.x;
  if (i < n) xh[i] = (__bf16)x[i];
}

// ---------------------------------------------------------------------------
// Kernel 1b: convert + transpose weights: Wt[n][k] = (bf16)W[k][n]
// ---------------------------------------------------------------------------
__global__ void k_cvt_w(const float* __restrict__ wq, const float* __restrict__ wk,
                        const float* __restrict__ wv,
                        __bf16* __restrict__ wqt, __bf16* __restrict__ wkt,
                        __bf16* __restrict__ wvt) {
  const float* w = (blockIdx.z == 0) ? wq : (blockIdx.z == 1) ? wk : wv;
  __bf16* wt     = (blockIdx.z == 0) ? wqt : (blockIdx.z == 1) ? wkt : wvt;
  int idx = blockIdx.x * blockDim.x + threadIdx.x;      // 0 .. 1M-1
  int k = idx >> 10;
  int n = idx & 1023;
  wt[(size_t)n * D_MODEL + k] = (__bf16)w[(size_t)k * DKDIM + n];
}

// ---------------------------------------------------------------------------
// Kernel 2: QKV projection GEMM.  C[m][n] = sum_k xh[m][k]*Wt[n][k] + bias[n].
// 256 threads = 8 waves; each wave computes a 32x64 tile (MT=2 x 4 n-tiles),
// amortizing the shared B-tile loads over twice the WMMAs (~22 FLOP/B vs L2).
// blockIdx.z: 0=Q (row-major bf16), 1=K (row-major bf16), 2=V (transposed).
// ---------------------------------------------------------------------------
__global__ __launch_bounds__(256)
void k_qkv_gemm(const __bf16* __restrict__ xh,
                const __bf16* __restrict__ wqt, const __bf16* __restrict__ wkt,
                const __bf16* __restrict__ wvt,
                const float* __restrict__ bq, const float* __restrict__ bk,
                const float* __restrict__ bv,
                __bf16* __restrict__ qh, __bf16* __restrict__ kh,
                __bf16* __restrict__ vt) {
  const int which = blockIdx.z;
  const __bf16* wt  = (which == 0) ? wqt : (which == 1) ? wkt : wvt;
  const float* bias = (which == 0) ? bq  : (which == 1) ? bk  : bv;

  const int wave = threadIdx.x >> 5;
  const int lane = threadIdx.x & 31;
  const int r  = lane & 15;     // row within A / col within B,C,D fragments
  const int hi = lane >> 4;     // half-wave selector

  const int m0 = blockIdx.y * 256 + wave * 32;   // 32 rows per wave (MT=2)
  const int n0 = blockIdx.x * 64;

  v8f acc[2][4];
#pragma unroll
  for (int i = 0; i < 2; ++i)
#pragma unroll
    for (int t = 0; t < 4; ++t) zero8(acc[i][t]);

  gemm_pingpong<2>(xh + (size_t)(m0 + r) * D_MODEL, (size_t)16 * D_MODEL,
                   wt + (size_t)(n0 + r) * D_MODEL, (size_t)16 * D_MODEL,
                   hi, D_MODEL, acc);

#pragma unroll
  for (int i = 0; i < 2; ++i) {
#pragma unroll
    for (int t = 0; t < 4; ++t) {
      const int col = n0 + t * 16 + r;     // D tile: lane col = r, row = j + 8*hi
      const float bsc = bias[col];
      if (which < 2) {
        __bf16* outp = (which == 0) ? qh : kh;
#pragma unroll
        for (int j = 0; j < 8; ++j) {
          int m = m0 + i * 16 + j + 8 * hi;
          outp[(size_t)m * DKDIM + col] = (__bf16)(acc[i][t][j] + bsc);
        }
      } else {
        // V stored transposed: vt[col][m] — contiguous over j -> one b128 store.
        __bf16* dst = vt + (size_t)col * MTOT + m0 + i * 16 + 8 * hi;
        v8bf pk;
#pragma unroll
        for (int j = 0; j < 8; ++j) pk[j] = (__bf16)(acc[i][t][j] + bsc);
        *(v8bf*)dst = pk;
      }
    }
  }
}

// ---------------------------------------------------------------------------
// Kernel 3: attention for one (batch, 16-query tile). 512 threads = 16 waves.
// Dynamic LDS: 16 x 2048 fp32 scores/probs = 128 KB (WGP has 320 KB).
// ---------------------------------------------------------------------------
__global__ __launch_bounds__(512)
void k_attn(const __bf16* __restrict__ qh, const __bf16* __restrict__ kh,
            const __bf16* __restrict__ vt, float* __restrict__ out) {
  extern __shared__ float sprob[];        // [16][SEQ]

  const int b    = blockIdx.y;
  const int q0   = blockIdx.x * 16;
  const int wave = threadIdx.x >> 5;
  const int lane = threadIdx.x & 31;
  const int r  = lane & 15;
  const int hi = lane >> 4;
  const size_t mq = (size_t)b * SEQ + q0;
  const float scale = 0.03125f;           // 1/sqrt(1024)

  // ---- Phase 1: scores = Q·Kᵀ. Two passes; wave covers 64 keys per pass. ---
  const __bf16* qrow = qh + (mq + r) * (size_t)DKDIM;
#pragma unroll
  for (int kb = 0; kb < 2; ++kb) {
    const int key0 = kb * 1024 + wave * 64;
    v8f acc[1][4];
#pragma unroll
    for (int t = 0; t < 4; ++t) zero8(acc[0][t]);

    // B = Kᵀ: lane col = key; fragment elems = K[key][k0 + hi*8 ..], K[key][k0 + 16 + hi*8 ..]
    gemm_pingpong<1>(qrow, 0,
                     kh + ((size_t)b * SEQ + key0 + r) * DKDIM,
                     (size_t)16 * DKDIM, hi, DKDIM, acc);
#pragma unroll
    for (int t = 0; t < 4; ++t) {
#pragma unroll
      for (int j = 0; j < 8; ++j)
        sprob[(j + 8 * hi) * SEQ + key0 + t * 16 + r] = acc[0][t][j] * scale;
    }
  }
  __syncthreads();

  // ---- Phase 2: softmax. Wave w owns query row w (16 rows, 16 waves). ----
  {
    float* row = sprob + wave * SEQ;
    float m = -3.0e38f;
    for (int i = lane; i < SEQ; i += 32) m = fmaxf(m, row[i]);
#pragma unroll
    for (int o = 16; o > 0; o >>= 1) m = fmaxf(m, __shfl_xor(m, o, 32));
    float sum = 0.0f;
    for (int i = lane; i < SEQ; i += 32) {
      float e = __expf(row[i] - m);
      row[i] = e;
      sum += e;
    }
#pragma unroll
    for (int o = 16; o > 0; o >>= 1) sum += __shfl_xor(sum, o, 32);
    const float inv = 1.0f / sum;
    for (int i = lane; i < SEQ; i += 32) row[i] *= inv;
  }
  __syncthreads();

  // ---- Phase 3: out = P·V. Wave w owns dv columns [w*64, w*64+64). ----
  {
    v8f acc[4];
#pragma unroll
    for (int t = 0; t < 4; ++t) zero8(acc[t]);

    const int n0 = wave * 64;
    // B = V: lane col = dv; fragment elems = V[key..][col] = Vt[col][key..]
    gemm_pv_pingpong(sprob + r * SEQ,
                     vt + (size_t)(n0 + r) * MTOT + (size_t)b * SEQ,
                     (size_t)16 * MTOT, hi, SEQ, acc);

#pragma unroll
    for (int t = 0; t < 4; ++t) {
      const int col = n0 + t * 16 + r;
#pragma unroll
      for (int j = 0; j < 8; ++j)
        out[(mq + j + 8 * hi) * (size_t)DVDIM + col] = acc[t][j];
    }
  }
}

// ---------------------------------------------------------------------------
// Launch. Workspace layout (bytes):
//   xh  @ 0          : 16384*1024*2 = 33554432
//   qh  @ 33554432   : 33554432
//   kh  @ 67108864   : 33554432
//   vt  @ 100663296  : 33554432   (transposed: [DV][MTOT])
//   wqt @ 134217728  : 2097152
//   wkt @ 136314880  : 2097152
//   wvt @ 138412032  : 2097152    (total 140509184 bytes)
// ---------------------------------------------------------------------------
extern "C" void kernel_launch(void* const* d_in, const int* in_sizes, int n_in,
                              void* d_out, int out_size, void* d_ws, size_t ws_size,
                              hipStream_t stream) {
  (void)in_sizes; (void)n_in; (void)out_size; (void)ws_size;
  const float* x  = (const float*)d_in[0];
  const float* Wq = (const float*)d_in[1];
  const float* bq = (const float*)d_in[2];
  const float* Wk = (const float*)d_in[3];
  const float* bk = (const float*)d_in[4];
  const float* Wv = (const float*)d_in[5];
  const float* bv = (const float*)d_in[6];

  char* ws = (char*)d_ws;
  __bf16* xh  = (__bf16*)(ws + 0);
  __bf16* qh  = (__bf16*)(ws + 33554432);
  __bf16* kh  = (__bf16*)(ws + 67108864);
  __bf16* vt  = (__bf16*)(ws + 100663296);
  __bf16* wqt = (__bf16*)(ws + 134217728);
  __bf16* wkt = (__bf16*)(ws + 136314880);
  __bf16* wvt = (__bf16*)(ws + 138412032);

  // 1) Precision conversion (+ weight transpose for contiguous B fragments).
  k_cvt_x<<<(MTOT * D_MODEL) / 256, 256, 0, stream>>>(x, xh, MTOT * D_MODEL);
  k_cvt_w<<<dim3((D_MODEL * DKDIM) / 256, 1, 3), 256, 0, stream>>>(
      Wq, Wk, Wv, wqt, wkt, wvt);

  // 2) QKV projections: grid (N/64, M/256, 3), 8 waves x (32x64) per block.
  k_qkv_gemm<<<dim3(DKDIM / 64, MTOT / 256, 3), 256, 0, stream>>>(
      xh, wqt, wkt, wvt, bq, bk, bv, qh, kh, vt);

  // 3) Attention: one block per (16-query tile, batch), 16 waves, 128 KB LDS.
  k_attn<<<dim3(SEQ / 16, BATCH), 512, 16 * SEQ * sizeof(float), stream>>>(
      qh, kh, vt, (float*)d_out);
}
